// MultiHeadAttention_58583353917610
// MI455X (gfx1250) — matrix operand order repaired and checked by reading, not compiled, not run on hardware
//
#include <hip/hip_runtime.h>
#include <hip/hip_bf16.h>

// ---------------------------------------------------------------------------
// MI455X (gfx1250) fused multi-head attention, f16 WMMA path, f32 accumulate.
// ---------------------------------------------------------------------------

#define D_MODEL 1024
#define N_HEADS 16
#define DK      64
#define SEQ     2048
#define BATCH   2
#define MROWS   (BATCH * SEQ)   // 4096

typedef _Float16 f16;
typedef __attribute__((ext_vector_type(16))) _Float16 v16h;
typedef __attribute__((ext_vector_type(8)))  float    v8f;
typedef __attribute__((ext_vector_type(4)))  unsigned int v4u;
typedef __attribute__((ext_vector_type(8)))  int          v8i;
typedef __attribute__((ext_vector_type(4)))  int          v4i;

union HFrag {
    v16h v;
    uint4 q[2];
    f16 h[16];
};

static __device__ __forceinline__ v8f vzero8() {
    v8f v;
#pragma unroll
    for (int i = 0; i < 8; ++i) v[i] = 0.0f;
    return v;
}

// Hard scheduling boundary: no instruction of any class may be reordered
// across this point. Used between LDS fragment-load groups and the WMMA
// chain so all ds_loads issue first and their latency overlaps the XDL pipe.
static __device__ __forceinline__ void sched_split() {
    __builtin_amdgcn_sched_barrier(0);
}

// gfx1250 async global->LDS copy (ASYNCcnt-tracked), CDNA5 ISA.
static __device__ __forceinline__ void async_copy_b128(unsigned lds_off, const void* gaddr) {
    asm volatile("global_load_async_to_lds_b128 %0, %1, off"
                 :: "v"(lds_off), "v"(gaddr)
                 : "memory");
}
static __device__ __forceinline__ void wait_asynccnt0() {
    asm volatile("s_wait_asynccnt 0x0" ::: "memory");
}

// DPP16 cross-lane ops for 16-lane row reductions (no LDS traffic).
template<int CTRL>
static __device__ __forceinline__ float dpp_permf(float x) {
    int i = __float_as_int(x);
    i = __builtin_amdgcn_update_dpp(i, i, CTRL, 0xF, 0xF, true);
    return __int_as_float(i);
}
static __device__ __forceinline__ float rowmax16(float x) {
    x = fmaxf(x, dpp_permf<0xB1>(x));   // quad_perm(1,0,3,2)  : xor 1
    x = fmaxf(x, dpp_permf<0x4E>(x));   // quad_perm(2,3,0,1)  : xor 2
    x = fmaxf(x, dpp_permf<0x141>(x));  // row_half_mirror     : xor 7 (~xor 4)
    x = fmaxf(x, dpp_permf<0x140>(x));  // row_mirror          : xor 15 (~xor 8)
    return x;
}
static __device__ __forceinline__ float rowsum16(float x) {
    x += dpp_permf<0xB1>(x);
    x += dpp_permf<0x4E>(x);
    x += dpp_permf<0x141>(x);
    x += dpp_permf<0x140>(x);
    return x;
}

// Tensor Data Mover: 2-D tile (f16 elements) global -> LDS, one wave issues.
// D# layout per CDNA5 ISA: group0 {count, lds_addr, global_addr,
// type=2}, group1 {data_size=2B, tensor_dim0/1, tile_dim0/1, dim0 stride}.
static __device__ __forceinline__ void tdm_load_2d_to_lds(
    unsigned lds_addr, const void* gsrc,
    int tile_w, int tile_h, int row_stride_elems)
{
    const unsigned long long ga = (unsigned long long)(size_t)gsrc;
    v4u g0;
    g0[0] = 1u;                                               // count=1 (valid D#)
    g0[1] = lds_addr;                                         // lds_addr [63:32]
    g0[2] = (unsigned)ga;                                     // global_addr[31:0]
    g0[3] = (unsigned)((ga >> 32) & 0x01FFFFFFull) | (2u << 30); // addr[56:32] | type=2
    v8i g1;
    g1[0] = (1 << 16);                                        // data_size=1 (2 bytes)
    g1[1] = (tile_w & 0xFFFF) << 16;                          // tensor_dim0[15:0]
    g1[2] = ((unsigned)tile_w >> 16) | ((tile_h & 0xFFFF) << 16); // dim0 hi | tensor_dim1 lo
    g1[3] = ((unsigned)tile_h >> 16) | ((tile_w & 0xFFFF) << 16); // dim1 hi | tile_dim0
    g1[4] = (tile_h & 0xFFFF);                                // tile_dim1 (tile_dim2=0)
    g1[5] = row_stride_elems;                                 // tensor_dim0_stride[31:0]
    g1[6] = 0;
    g1[7] = 0;
    v4i gz; gz[0] = gz[1] = gz[2] = gz[3] = 0;
#if defined(__clang_major__) && (__clang_major__ >= 23)
    v8i gz8;
#pragma unroll
    for (int i = 0; i < 8; ++i) gz8[i] = 0;
    __builtin_amdgcn_tensor_load_to_lds(g0, g1, gz, gz, gz8, 0);
#else
    __builtin_amdgcn_tensor_load_to_lds(g0, g1, gz, gz, 0);
#endif
}

// ---------------------------------------------------------------------------
// GEMM: Y[M,N] = X[M,K] * W[N,K]^T + bias[N]
// fp32 inputs converted to f16 while staging; double-buffered LDS ping-pong,
// one barrier per K-step, next tile's global loads issued before the WMMAs.
// Workgroup tile 128(M) x 128(N); 8 waves, each 32x64; K-step = 32.
// ---------------------------------------------------------------------------
template<bool F16OUT>
__global__ __launch_bounds__(256)
void gemm_xwt_kernel(const float* __restrict__ X,
                     const float* __restrict__ W,
                     const float* __restrict__ bias,
                     f16*   __restrict__ Yh,
                     float* __restrict__ Yf,
                     int M, int N, int K)
{
    __shared__ __align__(16) f16 sX [2][128 * 32];   // [m][k], pitch 32  (16 KB)
    __shared__ __align__(16) f16 sWt[2][32 * 128];   // [k][n], pitch 128 (16 KB)

    const int tid  = threadIdx.x;
    const int lane = tid & 31;
    const int wave = tid >> 5;
    const int l16  = lane & 15;
    const int lh   = lane >> 4;
    const int klo  = lh * 8;
    const int mw   = wave & 3;        // 4 M sub-tiles of 32
    const int nw   = wave >> 2;       // 2 N sub-tiles of 64

    const int mbase = blockIdx.y * 128;
    const int nbase = blockIdx.x * 128;

    const int xrow = tid >> 1, xseg = tid & 1;
    const int wn   = tid >> 1, wks  = tid & 1;

    const float* xsrc0 = X + (size_t)(mbase + xrow) * K + xseg * 16;
    const float* wsrc0 = W + (size_t)(nbase + wn)   * K + wks  * 16;

    float4 xf[4], wf[4];
    auto load_tiles = [&](int kb) {
        const float4* xs = (const float4*)(xsrc0 + kb);
        const float4* ws = (const float4*)(wsrc0 + kb);
#pragma unroll
        for (int j = 0; j < 4; ++j) { xf[j] = xs[j]; wf[j] = ws[j]; }
    };
    auto store_tiles = [&](int buf) {
        HFrag u;
#pragma unroll
        for (int j = 0; j < 4; ++j) {
            u.h[4*j+0] = (f16)xf[j].x; u.h[4*j+1] = (f16)xf[j].y;
            u.h[4*j+2] = (f16)xf[j].z; u.h[4*j+3] = (f16)xf[j].w;
        }
        uint4* dst = (uint4*)(&sX[buf][xrow * 32 + xseg * 16]);
        dst[0] = u.q[0];
        dst[1] = u.q[1];
#pragma unroll
        for (int j = 0; j < 4; ++j) {
            const int k0 = wks * 16 + j * 4;
            sWt[buf][(k0 + 0) * 128 + wn] = (f16)wf[j].x;
            sWt[buf][(k0 + 1) * 128 + wn] = (f16)wf[j].y;
            sWt[buf][(k0 + 2) * 128 + wn] = (f16)wf[j].z;
            sWt[buf][(k0 + 3) * 128 + wn] = (f16)wf[j].w;
        }
    };

    v8f acc[2][4];
#pragma unroll
    for (int s = 0; s < 2; ++s)
#pragma unroll
        for (int u = 0; u < 4; ++u) acc[s][u] = vzero8();

    load_tiles(0);
    store_tiles(0);
    __syncthreads();

    int p = 0;
    for (int kb = 0; kb < K; kb += 32) {
        const bool more = (kb + 32) < K;
        if (more) load_tiles(kb + 32);                        // overlap with WMMAs
        if (kb + 64 < K) {
            __builtin_prefetch(xsrc0 + kb + 64, 0, 1);
            __builtin_prefetch(wsrc0 + kb + 64, 0, 1);
        }

        HFrag afr[2], bfr[4];
#pragma unroll
        for (int s = 0; s < 2; ++s) {
            const f16* ptr = &sX[p][(mw * 32 + s * 16 + l16) * 32 + klo];
            afr[s].q[0] = *(const uint4*)ptr;
            afr[s].q[1] = *(const uint4*)(ptr + 16);
        }
#pragma unroll
        for (int u = 0; u < 4; ++u) {
            const f16* ptr = &sWt[p][nw * 64 + u * 16 + l16];
#pragma unroll
            for (int s = 0; s < 16; ++s)
                bfr[u].h[s] = ptr[(klo + (s & 7) + ((s >> 3) << 4)) * 128];
        }
        sched_split();   // all A/B fragment ds_loads issue before the first WMMA

#pragma unroll
        for (int s = 0; s < 2; ++s)
#pragma unroll
            for (int u = 0; u < 4; ++u)
                acc[s][u] = __builtin_amdgcn_wmma_f32_16x16x32_f16(
                    false, afr[s].v, false, bfr[u].v, (short)0, acc[s][u], false, false);

        if (more) store_tiles(p ^ 1);                         // fill back buffer
        __syncthreads();
        p ^= 1;
    }

#pragma unroll
    for (int s = 0; s < 2; ++s)
#pragma unroll
        for (int u = 0; u < 4; ++u) {
            const int n = nbase + nw * 64 + u * 16 + l16;
            const float bv = bias[n];
#pragma unroll
            for (int r = 0; r < 8; ++r) {
                const int m = mbase + mw * 32 + s * 16 + r + 8 * lh;
                const float y = acc[s][u][r] + bv;
                if constexpr (F16OUT) Yh[(size_t)m * N + n] = (f16)y;
                else                  Yf[(size_t)m * N + n] = y;
            }
        }
}

// ---------------------------------------------------------------------------
// Flash attention: 1 block = (b, h, 128 query rows); 8 waves x 16 rows each.
// Q tile loaded by the Tensor Data Mover; KV tiles double-buffered (V via async
// global->LDS DMA, K via registers transposed into LDS); DPP16 softmax reductions; all matmuls via WMMA f16.
// ---------------------------------------------------------------------------
__global__ __launch_bounds__(256)
void attn_kernel(const f16* __restrict__ qb,
                 const f16* __restrict__ kb,
                 const f16* __restrict__ vb,
                 float* __restrict__ ctx)
{
    __shared__ __align__(16) f16 sQ [128 * 64];     // TDM-loaded Q tile   (16 KB)
    __shared__ __align__(16) f16 sKt[2][64 * 32];   // [dk][kv] transposed ( 8 KB)
    __shared__ __align__(16) f16 sV [2][32 * 64];   // [kv][dk]            ( 8 KB)
    __shared__ __align__(16) f16 sP [8][16 * 32];   // per-wave P patch    ( 8 KB)

    const int tid  = threadIdx.x;
    const int lane = tid & 31;
    const int wave = tid >> 5;
    const int l16  = lane & 15;
    const int lh   = lane >> 4;
    const int klo  = lh * 8;

    const int bh = blockIdx.y;
    const int b  = bh >> 4;
    const int h  = bh & 15;
    const int qbase = blockIdx.x * 128;

    const size_t headoff = (size_t)h * DK;
    const size_t browoff = (size_t)b * SEQ;

    const int skv = tid >> 3;    // 0..31 kv row staged by this thread
    const int sg8 = tid & 7;     // 8-element (16 B) segment within a row

    HFrag kreg;
    auto stage_v_async = [&](int buf, int kv0_) {
        const f16* src = vb + (browoff + kv0_ + skv) * (size_t)D_MODEL + headoff + sg8 * 8;
        async_copy_b128((unsigned)(size_t)(&sV[buf][skv * 64 + sg8 * 8]), src);
    };
    auto load_k = [&](int kv0_) {
        const f16* src = kb + (browoff + kv0_ + skv) * (size_t)D_MODEL + headoff + sg8 * 8;
        kreg.q[0] = *(const uint4*)src;
    };
    auto store_k = [&](int buf) {
#pragma unroll
        for (int i = 0; i < 8; ++i)
            sKt[buf][(sg8 * 8 + i) * 32 + skv] = kreg.h[i];
    };

    // ---- prologue: TDM the Q tile, stage KV tile 0 (V async, K via registers) ----
    if (wave == 0) {
        tdm_load_2d_to_lds((unsigned)(size_t)(&sQ[0]),
                           qb + (browoff + qbase) * (size_t)D_MODEL + headoff,
                           DK, 128, D_MODEL);
        __builtin_amdgcn_s_wait_tensorcnt(0);
    }
    stage_v_async(0, 0);
    load_k(0);
    store_k(0);
    wait_asynccnt0();
    __syncthreads();

    // ---- this wave's Q A-fragments from LDS (dk=64 -> 2 K-chunks) ----
    HFrag aq[2];
    {
        const f16* qp = sQ + (wave * 16 + l16) * 64;
#pragma unroll
        for (int c = 0; c < 2; ++c) {
            aq[c].q[0] = *(const uint4*)(qp + c * 32 + klo);
            aq[c].q[1] = *(const uint4*)(qp + c * 32 + klo + 16);
        }
    }

    v8f mrow, lrow, acc[4];
#pragma unroll
    for (int r = 0; r < 8; ++r) { mrow[r] = -1e30f; lrow[r] = 0.0f; }
#pragma unroll
    for (int f = 0; f < 4; ++f) acc[f] = vzero8();

    const float scale = 0.125f;  // 1/sqrt(64)

    int p = 0;
    for (int kv0 = 0; kv0 < SEQ; kv0 += 32) {
        const bool more = (kv0 + 32) < SEQ;
        if (more) { stage_v_async(p ^ 1, kv0 + 32); load_k(kv0 + 32); }
        if (kv0 + 64 < SEQ) {
            __builtin_prefetch(kb + (browoff + kv0 + 64 + skv) * (size_t)D_MODEL + headoff, 0, 1);
            __builtin_prefetch(vb + (browoff + kv0 + 64 + skv) * (size_t)D_MODEL + headoff, 0, 1);
        }

        // ---- scores S = Q * K^T : batch all K-fragment ds_loads, then WMMA chain ----
        HFrag bk[2][2];
#pragma unroll
        for (int c = 0; c < 2; ++c)
#pragma unroll
            for (int nt = 0; nt < 2; ++nt) {
                const f16* ptr = &sKt[p][(c * 32) * 32 + nt * 16 + l16];
#pragma unroll
                for (int s = 0; s < 16; ++s)
                    bk[c][nt].h[s] = ptr[(klo + (s & 7) + ((s >> 3) << 4)) * 32];
            }
        sched_split();   // all K-fragment ds_loads issue before the first WMMA

        v8f sc[2];
        sc[0] = vzero8(); sc[1] = vzero8();
#pragma unroll
        for (int nt = 0; nt < 2; ++nt)
#pragma unroll
            for (int c = 0; c < 2; ++c)
                sc[nt] = __builtin_amdgcn_wmma_f32_16x16x32_f16(
                    false, aq[c].v, false, bk[c][nt].v, (short)0, sc[nt], false, false);

        // ---- online softmax: DPP16 row reductions, no LDS traffic ----
        v8f alpha;
#pragma unroll
        for (int r = 0; r < 8; ++r) {
            sc[0][r] *= scale;
            sc[1][r] *= scale;
            const float cm = rowmax16(fmaxf(sc[0][r], sc[1][r]));
            const float mn = fmaxf(mrow[r], cm);
            alpha[r] = __expf(mrow[r] - mn);
            mrow[r]  = mn;
            sc[0][r] = __expf(sc[0][r] - mn);
            sc[1][r] = __expf(sc[1][r] - mn);
            const float rs = rowsum16(sc[0][r] + sc[1][r]);
            lrow[r] = lrow[r] * alpha[r] + rs;
        }
#pragma unroll
        for (int f = 0; f < 4; ++f)
#pragma unroll
            for (int r = 0; r < 8; ++r)
                acc[f][r] *= alpha[r];

        // ---- V B-frags issued first (independent of the P patch) ----
        HFrag bvf[4];
#pragma unroll
        for (int f = 0; f < 4; ++f) {
            const f16* ptr = &sV[p][f * 16 + l16];
#pragma unroll
            for (int s = 0; s < 16; ++s)
                bvf[f].h[s] = ptr[(klo + (s & 7) + ((s >> 3) << 4)) * 64];
        }
        sched_split();   // V fragments in flight before the P patch round-trip

        // ---- transpose P (D-layout -> A-layout) via per-wave LDS patch ----
        f16* pw = sP[wave];
#pragma unroll
        for (int nt = 0; nt < 2; ++nt)
#pragma unroll
            for (int r = 0; r < 8; ++r)
                pw[(r + 8 * lh) * 32 + nt * 16 + l16] = (f16)sc[nt][r];

        HFrag ap;
        {
            const f16* pr = pw + l16 * 32 + klo;
            ap.q[0] = *(const uint4*)pr;
            ap.q[1] = *(const uint4*)(pr + 16);
        }

        // ---- ctx += P * V ----
#pragma unroll
        for (int f = 0; f < 4; ++f)
            acc[f] = __builtin_amdgcn_wmma_f32_16x16x32_f16(
                false, ap.v, false, bvf[f].v, (short)0, acc[f], false, false);

        if (more) store_k(p ^ 1);   // scatter next K tile into back buffer
        wait_asynccnt0();           // next V tile must land before barrier
        __syncthreads();
        p ^= 1;
    }

    // ---- epilogue: normalize and scatter ctx (f32, [B,S,D] head slice) ----
#pragma unroll
    for (int f = 0; f < 4; ++f) {
#pragma unroll
        for (int r = 0; r < 8; ++r) {
            const int m = qbase + wave * 16 + r + 8 * lh;
            const float v = acc[f][r] / lrow[r];
            ctx[(browoff + m) * (size_t)D_MODEL + headoff + f * 16 + l16] = v;
        }
    }
}

// ---------------------------------------------------------------------------
// Host launcher
// ---------------------------------------------------------------------------
extern "C" void kernel_launch(void* const* d_in, const int* in_sizes, int n_in,
                              void* d_out, int out_size, void* d_ws, size_t ws_size,
                              hipStream_t stream)
{
    (void)in_sizes; (void)n_in; (void)out_size; (void)ws_size;

    const float* Q  = (const float*)d_in[0];
    const float* K  = (const float*)d_in[1];
    const float* V  = (const float*)d_in[2];
    const float* Wq = (const float*)d_in[3];
    const float* bq = (const float*)d_in[4];
    const float* Wk = (const float*)d_in[5];
    const float* bk = (const float*)d_in[6];
    const float* Wv = (const float*)d_in[7];
    const float* bv = (const float*)d_in[8];
    const float* Wo = (const float*)d_in[9];
    const float* bo = (const float*)d_in[10];

    // Workspace: qbuf f16 8MB | kbuf f16 8MB | vbuf f16 8MB | ctx f32 16MB
    char* wsb = (char*)d_ws;
    const size_t HBYTES = (size_t)MROWS * D_MODEL * 2;
    f16*   qbuf   = (f16*)(wsb);
    f16*   kbuf   = (f16*)(wsb + HBYTES);
    f16*   vbuf   = (f16*)(wsb + 2 * HBYTES);
    float* ctxbuf = (float*)(wsb + 3 * HBYTES);

    const dim3 ggrid(D_MODEL / 128, MROWS / 128);  // (8, 32)

    gemm_xwt_kernel<true ><<<ggrid, 256, 0, stream>>>(Q, Wq, bq, qbuf, nullptr, MROWS, D_MODEL, D_MODEL);
    gemm_xwt_kernel<true ><<<ggrid, 256, 0, stream>>>(K, Wk, bk, kbuf, nullptr, MROWS, D_MODEL, D_MODEL);
    gemm_xwt_kernel<true ><<<ggrid, 256, 0, stream>>>(V, Wv, bv, vbuf, nullptr, MROWS, D_MODEL, D_MODEL);

    attn_kernel<<<dim3(SEQ / 128, BATCH * N_HEADS), 256, 0, stream>>>(qbuf, kbuf, vbuf, ctxbuf);

    gemm_xwt_kernel<false><<<ggrid, 256, 0, stream>>>(ctxbuf, Wo, bo, nullptr, (float*)d_out, MROWS, D_MODEL, D_MODEL);
}
